// SCLDecoderHY_22686017257979
// MI455X (gfx1250) — matrix operand order repaired and checked by reading, not compiled, not run on hardware
//
#include <hip/hip_runtime.h>

typedef __bf16 bf16_t;
typedef __attribute__((ext_vector_type(16))) __bf16 v16bf;
typedef __attribute__((ext_vector_type(8)))  __bf16 v8bf;
typedef __attribute__((ext_vector_type(8)))  float  v8f;
typedef __attribute__((ext_vector_type(4)))  unsigned int v4u;
typedef __attribute__((ext_vector_type(8)))  int v8i;
typedef __attribute__((ext_vector_type(4)))  int v4i;

#if __has_builtin(__builtin_amdgcn_tensor_load_to_lds) && __has_builtin(__builtin_amdgcn_s_wait_tensorcnt)
#define USE_TDM 1
#else
#define USE_TDM 0
#endif

#define Bn 64
#define Ln 8
#define Nn 128
#define Dn 128
#define BLn (Bn * Ln)   // 512

// xbuf element offset for depth d (layout: (B, L, 128>>d) per depth, contiguous)
__host__ __device__ inline long xoffE(int d) { return (long)BLn * (256 - (256 >> d)); }

// WMMA fragment-major swizzle for a (K,128) weight matrix:
// element (k,n) -> ((kt*8 + t)*32 + lane)*16 + j  with kt=k>>5, hi=(k>>4)&1,
// j=k&15, t=n>>4, lane=hi*16+(n&15). Lane's 16 B-fragment elems are contiguous.
__device__ inline long wswz(int k, int n) {
  int kt = k >> 5, hi = (k >> 4) & 1, j = k & 15;
  int t = n >> 4, lane = hi * 16 + (n & 15);
  return (((long)kt * 8 + t) * 32 + lane) * 16 + j;
}

// ---------------------------------------------------------------------------
// Init: bf16-convert + fragment-swizzle weights, depth-0 decoder embeddings, pm
// ---------------------------------------------------------------------------
__global__ void init_kernel(const float* ecn, const float* dcn, const float* ebn,
                            const float* dbn, const int* y, const float* dec_obs,
                            bf16_t* wbf, float* ech0, float* pm) {
  long tid = (long)blockIdx.x * blockDim.x + threadIdx.x;
  long stride = (long)gridDim.x * blockDim.x;
  for (long i = tid; i < 32768; i += stride) {
    int k = (int)(i >> 7), n = (int)(i & 127);
    wbf[wswz(k, n)] = (bf16_t)ecn[i];
  }
  for (long i = tid; i < 32768; i += stride) {
    int k = (int)(i >> 7), n = (int)(i & 127);
    wbf[32768 + wswz(k, n)] = (bf16_t)dcn[i];
  }
  for (long i = tid; i < 49152; i += stride) {
    int k = (int)(i >> 7), n = (int)(i & 127);
    wbf[65536 + wswz(k, n)] = (bf16_t)ebn[i];
  }
  for (long i = tid; i < 49152; i += stride) {
    int k = (int)(i >> 7), n = (int)(i & 127);
    wbf[114688 + wswz(k, n)] = (bf16_t)dbn[i];
  }
  for (long i = tid; i < (long)Bn * Nn * Dn; i += stride) {
    int k = (int)(i & 127);
    int pos = (int)((i >> 7) & 127);
    int b = (int)(i >> 14);
    ech0[i] = dec_obs[(long)y[b * Nn + pos] * Dn + k];
  }
  for (long i = tid; i < BLn; i += stride) pm[i] = (i & 7) ? 1e8f : 0.f;
}

__global__ void reset_comp_kernel(int* cb) { cb[threadIdx.x] = threadIdx.x & 7; }

// ---------------------------------------------------------------------------
// bf16 WMMA GEMM:  out[rows,128] = relu(A[rows,K] @ W[K,128] + bias)
// A rows are (b,l,i); features k<256 read e[b, l', 2i + (k>=128), k&127]
// with l' = comp[b,l] (list indirection), k>=256 read lab_emb[xb[b,l,i]].
// W is pre-swizzled fragment-major; optionally DMA'd to LDS via the TDM.
// ---------------------------------------------------------------------------
struct GemmParams {
  const float* A;
  long sB, sL, sP;          // strides (floats) for b, l, pos
  const int* comp;          // list indirection (nullptr = identity)
  const int* xb;            // hard-decision labels (bn only)
  const float* lab;         // label embedding (2,128) (bn only)
  const bf16_t* W;          // (K,128) bf16, fragment-major
  const float* bias;        // (128)
  float* out;               // (rows,128)
  int K;                    // 256 or 384 (host-side dispatch)
  int h;                    // output positions per (b,l)
};

template <int KK>
__global__ __launch_bounds__(128) void gemm_wmma_kernel(GemmParams p) {
  extern __shared__ char smem[];
  bf16_t (*As)[40] = (bf16_t(*)[40])smem;            // 64 x 40 bf16 (padded)
#if USE_TDM
  bf16_t* Wl = (bf16_t*)(smem + 8192);               // KK*128 bf16
#endif
  const int tid = threadIdx.x;
  const int wave = tid >> 5;
  const int lane = tid & 31;
  const int row0 = blockIdx.x * 64;
  const int hl = p.h;

  // Per-thread A staging coordinates (fixed across the K loop).
  const int rr = tid >> 1;                 // 0..63: tile row
  const int kh = (tid & 1) << 4;           // 0 or 16: K-chunk within tile
  int grow = row0 + rr;
  int b = grow / (Ln * hl);
  int rem = grow - b * Ln * hl;
  int l = rem / hl;
  int i = rem - l * hl;
  int ll = p.comp ? p.comp[b * Ln + l] : l;
  const float* pe = p.A + (long)b * p.sB + (long)ll * p.sL + (long)(2 * i) * p.sP;
  const float* po = pe + p.sP;
  const float* lb = p.xb ? (p.lab + (long)p.xb[(b * Ln + l) * hl + i] * Dn) : nullptr;

#if USE_TDM
  // Wave 0: DMA the whole fragment-major weight matrix into LDS via the TDM.
  if (tid < 32) {
    unsigned long long ga = (unsigned long long)(const void*)p.W;
    unsigned int ldsoff = (unsigned int)(unsigned long long)(void*)Wl;
    unsigned int nel = (unsigned int)(KK * 128);
    // D# group0: count=1 | lds_addr | global_addr[56:0] | type=2
    v4u g0 = {1u, ldsoff,
              (unsigned int)(ga & 0xFFFFFFFFu),
              (unsigned int)((ga >> 32) & 0x01FFFFFFu) | 0x80000000u};
    // D# group1: data_size=2B; 1-D tensor/tile of nel elements
    v8i g1 = {(int)(1u << 16),                       // data_size=1 (2 bytes)
              (int)((nel & 0xFFFFu) << 16),          // tensor_dim0 lo16
              (int)(((nel >> 16) & 0xFFFFu) | (1u << 16)),  // dim0 hi16 | tensor_dim1=1
              (int)(nel << 16),                      // tile_dim0
              0,                                     // tile_dim1/2 unused
              (int)nel,                              // tensor_dim0_stride lo32
              0, 0};
    v4i gz = {0, 0, 0, 0};
    v8i gz8 = {0, 0, 0, 0, 0, 0, 0, 0};
    __builtin_amdgcn_tensor_load_to_lds(g0, g1, gz, gz, gz8, 0);
    __builtin_amdgcn_s_wait_tensorcnt(0);
  }
  __syncthreads();
#endif

  const int mr = lane & 15;
  const int hig = lane >> 4;
  v8f acc[8] = {};
#pragma unroll
  for (int kt = 0; kt < (KK >> 5); ++kt) {
    const int k0 = kt << 5;
    // ---- stage A: 16 contiguous features per thread, f32 -> bf16 ----
    {
      int k = k0 + kh;
      const float* src;
      if (KK > 256 && k >= 256) src = lb + (k - 256);
      else src = (((k >> 7) & 1) ? po : pe) + (k & 127);
      const float4* s4 = (const float4*)src;
      float4 q0 = s4[0], q1 = s4[1], q2 = s4[2], q3 = s4[3];
      v8bf vlo, vhi;
      vlo[0] = (bf16_t)q0.x; vlo[1] = (bf16_t)q0.y;
      vlo[2] = (bf16_t)q0.z; vlo[3] = (bf16_t)q0.w;
      vlo[4] = (bf16_t)q1.x; vlo[5] = (bf16_t)q1.y;
      vlo[6] = (bf16_t)q1.z; vlo[7] = (bf16_t)q1.w;
      vhi[0] = (bf16_t)q2.x; vhi[1] = (bf16_t)q2.y;
      vhi[2] = (bf16_t)q2.z; vhi[3] = (bf16_t)q2.w;
      vhi[4] = (bf16_t)q3.x; vhi[5] = (bf16_t)q3.y;
      vhi[6] = (bf16_t)q3.z; vhi[7] = (bf16_t)q3.w;
      *(v8bf*)&As[rr][kh] = vlo;
      *(v8bf*)&As[rr][kh + 8] = vhi;
    }
    __syncthreads();
    // ---- A fragment: 2 x b128 LDS loads ----
    v8bf alo = *(const v8bf*)&As[wave * 16 + mr][hig * 16];
    v8bf ahi = *(const v8bf*)&As[wave * 16 + mr][hig * 16 + 8];
    v16bf a = __builtin_shufflevector(alo, ahi, 0, 1, 2, 3, 4, 5, 6, 7,
                                      8, 9, 10, 11, 12, 13, 14, 15);
#if !USE_TDM
    if (lane == 0 && kt + 1 < (KK >> 5))
      __builtin_prefetch(p.W + (((long)(kt + 1) * 8) * 32) * 16, 0, 1);
#endif
#pragma unroll
    for (int t = 0; t < 8; ++t) {
      long fidx = (((long)kt * 8 + t) * 32 + lane) * 16;   // 32B contiguous frag
#if USE_TDM
      v16bf wf = *(const v16bf*)&Wl[fidx];
#else
      v16bf wf = *(const v16bf*)(p.W + fidx);
#endif
      acc[t] = __builtin_amdgcn_wmma_f32_16x16x32_bf16(
          false, a, false, wf, (short)0, acc[t], false, false);
    }
    __syncthreads();
  }
  // ---- epilogue: C/D layout M = v + hig*8, N = lane&15 ----
#pragma unroll
  for (int t = 0; t < 8; ++t) {
#pragma unroll
    for (int v = 0; v < 8; ++v) {
      int orow = row0 + wave * 16 + hig * 8 + v;
      int col = t * 16 + mr;
      float val = acc[t][v] + p.bias[col];
      p.out[(long)orow * Dn + col] = val > 0.f ? val : 0.f;
    }
  }
}

// ---------------------------------------------------------------------------
// Leaf: softmax heads, LLR, frozen/info handling, 16-way top-k, pm & perm
// ---------------------------------------------------------------------------
__global__ __launch_bounds__(64) void leaf_kernel(
    const float* eco7, const float* ech7,
    const float* encW, const float* encB, const float* decW, const float* decB,
    const int* f, const float* r, float* pm, int* perm,
    int* uhat, float* pu_g, float* puy_g, int* xdst, int xstride, int c) {
  int b = blockIdx.x;
  int t = threadIdx.x;
  int l = t >> 3, part = t & 7;
  const float* co = eco7 + (long)(b * Ln + l) * Dn;
  const float* ch = ech7 + (long)(b * Ln + l) * Dn;
  float sE0 = 0.f, sE1 = 0.f, sD0 = 0.f, sD1 = 0.f;
  for (int k = part * 16; k < part * 16 + 16; ++k) {
    float vco = co[k], vch = ch[k];
    sE0 += vco * encW[k * 2 + 0];
    sE1 += vco * encW[k * 2 + 1];
    sD0 += vch * decW[k * 2 + 0];
    sD1 += vch * decW[k * 2 + 1];
  }
  for (int d = 4; d > 0; d >>= 1) {
    sE0 += __shfl_down(sE0, d, 8);
    sE1 += __shfl_down(sE1, d, 8);
    sD0 += __shfl_down(sD0, d, 8);
    sD1 += __shfl_down(sD1, d, 8);
  }
  __shared__ float aE0[Ln], aE1[Ln], aD0[Ln], aD1[Ln];
  if (part == 0) {
    aE0[l] = sE0 + encB[0]; aE1[l] = sE1 + encB[1];
    aD0[l] = sD0 + decB[0]; aD1[l] = sD1 + decB[1];
  }
  __syncthreads();
  if (t == 0) {
    float pu0[Ln], pu1[Ln], dm0[Ln], dm1[Ln], allr[Ln];
    int hd_[Ln];
    for (int j = 0; j < Ln; ++j) {
      float e0 = aE0[j], e1 = aE1[j];
      float m = fmaxf(e0, e1);
      float x0 = expf(e0 - m), x1 = expf(e1 - m);
      pu0[j] = x0 / (x0 + x1); pu1[j] = 1.f - pu0[j];
      float d0 = aD0[j], d1 = aD1[j];
      m = fmaxf(d0, d1);
      float y0 = expf(d0 - m), y1 = expf(d1 - m);
      float s = y0 + y1;
      dm0[j] = y0 / s; dm1[j] = y1 / s;
      float p1 = fminf(fmaxf(dm1[j], 1e-6f), 1.f - 1e-6f);
      float llr = logf(p1) - logf(1.f - p1);
      allr[j] = fabsf(llr);
      hd_[j] = dm1[j] > dm0[j] ? 1 : 0;
    }
    int fz = f[b * Nn + c];
    float rv = r[b * Nn + c];
    int xv[Ln];
    if (fz != 2) {                                  // frozen leaf
      for (int j = 0; j < Ln; ++j) {
        int fb = (rv > pu0[j]) ? 1 : 0;
        xv[j] = fb;
        if (hd_[j] != fb) pm[b * Ln + j] += allr[j];
        perm[b * Ln + j] = j;
      }
    } else {                                        // information leaf: top-k prune
      float pdup[16];
      for (int j = 0; j < Ln; ++j) {
        pdup[j] = pm[b * Ln + j];
        pdup[Ln + j] = pm[b * Ln + j] + allr[j];
      }
      bool used[16];
      for (int q = 0; q < 16; ++q) used[q] = false;
      int sel[Ln];
      for (int j = 0; j < Ln; ++j) {                // L smallest, lowest-index ties
        int bi = -1; float bv = 3.4e38f;
        for (int q = 0; q < 16; ++q)
          if (!used[q] && pdup[q] < bv) { bv = pdup[q]; bi = q; }
        used[bi] = true; sel[j] = bi;
      }
      for (int j = 1; j < Ln; ++j) {                // sort survivor indices ascending
        int key = sel[j], k2 = j - 1;
        while (k2 >= 0 && sel[k2] > key) { sel[k2 + 1] = sel[k2]; --k2; }
        sel[k2 + 1] = key;
      }
      for (int j = 0; j < Ln; ++j) {
        pm[b * Ln + j] = pdup[sel[j]];
        perm[b * Ln + j] = sel[j] & 7;
        xv[j] = (sel[j] < Ln) ? hd_[sel[j]] : 1 - hd_[sel[j] - Ln];
      }
    }
    for (int j = 0; j < Ln; ++j) {
      uhat[(b * Ln + j) * Nn + c] = xv[j];
      long o = ((long)(b * Ln + j) * Nn + c) * 2;
      pu_g[o + 0] = pu0[j];  pu_g[o + 1] = pu1[j];
      puy_g[o + 0] = dm0[j]; puy_g[o + 1] = dm1[j];
      xdst[(b * Ln + j) * xstride] = xv[j];
    }
  }
}

// ---------------------------------------------------------------------------
// Apply list permutation to decoded columns, pending partial sums, comp maps
// ---------------------------------------------------------------------------
__global__ void perm_apply_kernel(const int* perm, int* uhat, float* pu, float* puy,
                                  int* xb, int* comp, int c, int pendMask, int activeMask) {
  long tid0 = (long)blockIdx.x * blockDim.x + threadIdx.x;
  long stride = (long)gridDim.x * blockDim.x;
  for (long t = tid0; t < (long)Bn * c; t += stride) {
    int b = (int)(t / c), col = (int)(t % c);
    const int* pb = perm + b * Ln;
    int tu[Ln]; float t0[Ln], t1[Ln], t2[Ln], t3[Ln];
    for (int l = 0; l < Ln; ++l) {
      int s = pb[l];
      long o = ((long)(b * Ln + s) * Nn + col);
      tu[l] = uhat[o];
      t0[l] = pu[o * 2 + 0];  t1[l] = pu[o * 2 + 1];
      t2[l] = puy[o * 2 + 0]; t3[l] = puy[o * 2 + 1];
    }
    for (int l = 0; l < Ln; ++l) {
      long o = ((long)(b * Ln + l) * Nn + col);
      uhat[o] = tu[l];
      pu[o * 2 + 0] = t0[l];  pu[o * 2 + 1] = t1[l];
      puy[o * 2 + 0] = t2[l]; puy[o * 2 + 1] = t3[l];
    }
  }
  for (int dd = 1; dd <= 7; ++dd) {
    if (!(pendMask & (1 << dd))) continue;
    int n = 128 >> dd;
    int* base = xb + xoffE(dd);
    for (long t = tid0; t < (long)Bn * n; t += stride) {
      int b = (int)(t / n), i = (int)(t % n);
      int tmp[Ln];
      for (int l = 0; l < Ln; ++l) tmp[l] = base[(b * Ln + perm[b * Ln + l]) * n + i];
      for (int l = 0; l < Ln; ++l) base[(b * Ln + l) * n + i] = tmp[l];
    }
  }
  for (int d = 0; d < 7; ++d) {
    if (!(activeMask & (1 << d))) continue;
    int* cb = comp + d * BLn;
    for (long t = tid0; t < Bn; t += stride) {
      int b = (int)t; int tmp[Ln];
      for (int l = 0; l < Ln; ++l) tmp[l] = cb[b * Ln + perm[b * Ln + l]];
      for (int l = 0; l < Ln; ++l) cb[b * Ln + l] = tmp[l];
    }
  }
}

// ---------------------------------------------------------------------------
// Upward XOR merges of partial sums (m trailing-one levels)
// ---------------------------------------------------------------------------
__global__ __launch_bounds__(128) void merge_kernel(const int* xcur, int* xb, int m) {
  __shared__ int bufA[128], bufB[128];
  int bl = blockIdx.x, t = threadIdx.x;
  int* cur = bufA; int* nxt = bufB;
  if (t == 0) cur[0] = xcur[(long)bl * 128];
  __syncthreads();
  int w = 1;
  for (int k = 1; k <= m; ++k) {
    const int* left = xb + xoffE(8 - k) + (long)bl * w;
    if (t < w) {
      int Lx = left[t], Rx = cur[t];
      nxt[2 * t] = (Lx ^ Rx) & 1;
      nxt[2 * t + 1] = Rx;
    }
    __syncthreads();
    int* tmp = cur; cur = nxt; nxt = tmp;
    w <<= 1;
  }
  if (t < w) xb[xoffE(7 - m) + (long)bl * w + t] = cur[t];
}

// ---------------------------------------------------------------------------
// Final: pick best list per batch, emit (uhat, p_uy, p_u)
// ---------------------------------------------------------------------------
__global__ void final_kernel(const float* pm, const int* uhat, const float* puy,
                             const float* pu, float* out) {
  int b = blockIdx.x;
  __shared__ int sbi;
  if (threadIdx.x == 0) {
    float bv = pm[b * Ln]; int bi = 0;
    for (int l = 1; l < Ln; ++l)
      if (pm[b * Ln + l] < bv) { bv = pm[b * Ln + l]; bi = l; }
    sbi = bi;
  }
  __syncthreads();
  int bi = sbi;
  for (int col = threadIdx.x; col < Nn; col += blockDim.x) {
    long src = (long)(b * Ln + bi) * Nn + col;
    out[b * Nn + col] = (float)uhat[src];
    out[8192 + (long)(b * Nn + col) * 2 + 0] = puy[src * 2 + 0];
    out[8192 + (long)(b * Nn + col) * 2 + 1] = puy[src * 2 + 1];
    out[24576 + (long)(b * Nn + col) * 2 + 0] = pu[src * 2 + 0];
    out[24576 + (long)(b * Nn + col) * 2 + 1] = pu[src * 2 + 1];
  }
}

// ---------------------------------------------------------------------------
extern "C" void kernel_launch(void* const* d_in, const int* in_sizes, int n_in,
                              void* d_out, int out_size, void* d_ws, size_t ws_size,
                              hipStream_t stream) {
  (void)in_sizes; (void)n_in; (void)out_size; (void)ws_size;
  const int*   y        = (const int*)d_in[0];
  const int*   f        = (const int*)d_in[1];
  const float* r        = (const float*)d_in[2];
  const float* enc_obs  = (const float*)d_in[3];
  const float* dec_obs  = (const float*)d_in[4];
  const float* enc_lab  = (const float*)d_in[5];
  const float* dec_lab  = (const float*)d_in[6];
  const float* enc_llr_W = (const float*)d_in[7];
  const float* enc_llr_b = (const float*)d_in[8];
  const float* dec_llr_W = (const float*)d_in[9];
  const float* dec_llr_b = (const float*)d_in[10];
  const float* enc_cn_W = (const float*)d_in[11];
  const float* enc_cn_b = (const float*)d_in[12];
  const float* dec_cn_W = (const float*)d_in[13];
  const float* dec_cn_b = (const float*)d_in[14];
  const float* enc_bn_W = (const float*)d_in[15];
  const float* enc_bn_b = (const float*)d_in[16];
  const float* dec_bn_W = (const float*)d_in[17];
  const float* dec_bn_b = (const float*)d_in[18];

  char* ws = (char*)d_ws;
  size_t off = 0;
  auto alloc = [&](size_t bytes) -> void* {
    off = (off + 255) & ~(size_t)255;
    void* p = ws + off;
    off += bytes;
    return p;
  };
  bf16_t* wbf  = (bf16_t*)alloc((size_t)163840 * sizeof(bf16_t));
  float*  ech0 = (float*)alloc((size_t)Bn * Nn * Dn * 4);
  float*  eco[8] = {}; float* ech[8] = {};
  for (int d = 1; d <= 7; ++d) {
    size_t bytes = (size_t)BLn * (128 >> d) * Dn * 4;
    eco[d] = (float*)alloc(bytes);
    ech[d] = (float*)alloc(bytes);
  }
  int*   UHAT = (int*)alloc((size_t)BLn * Nn * 4);
  float* PU   = (float*)alloc((size_t)BLn * Nn * 2 * 4);
  float* PUY  = (float*)alloc((size_t)BLn * Nn * 2 * 4);
  float* PM   = (float*)alloc((size_t)BLn * 4);
  int*   PERM = (int*)alloc((size_t)BLn * 4);
  int*   COMP = (int*)alloc((size_t)7 * BLn * 4);
  int*   XBUF = (int*)alloc((size_t)BLn * 255 * 4);
  int*   XCUR = (int*)alloc((size_t)BLn * 128 * 4);

  init_kernel<<<256, 256, 0, stream>>>(enc_cn_W, dec_cn_W, enc_bn_W, dec_bn_W,
                                       y, dec_obs, wbf, ech0, PM);

  auto setA = [&](GemmParams& g, int d, bool enc) {
    if (d == 0) {
      if (enc) { g.A = enc_obs + 2 * Dn; g.sB = 0; g.sL = 0; g.sP = 0; }
      else     { g.A = ech0; g.sB = (long)Nn * Dn; g.sL = 0; g.sP = Dn; }
    } else {
      int n = 128 >> d;
      g.A = enc ? eco[d] : ech[d];
      g.sB = (long)Ln * n * Dn; g.sL = (long)n * Dn; g.sP = Dn;
    }
  };
  auto gemm = [&](const GemmParams& g) {
    int rows = BLn * g.h;
    size_t shbytes = 8192 + (size_t)g.K * 128 * 2;
    if (g.K == 256)
      gemm_wmma_kernel<256><<<dim3(rows / 64), dim3(128), shbytes, stream>>>(g);
    else
      gemm_wmma_kernel<384><<<dim3(rows / 64), dim3(128), shbytes, stream>>>(g);
  };
  auto cn = [&](int d) {
    reset_comp_kernel<<<1, BLn, 0, stream>>>(COMP + d * BLn);
    int h = (128 >> d) >> 1;
    for (int e = 0; e < 2; ++e) {
      bool enc = (e == 0);
      GemmParams g{};
      setA(g, d, enc);
      g.comp = nullptr; g.xb = nullptr; g.lab = nullptr;
      g.W = wbf + (enc ? 0 : 32768);
      g.bias = enc ? enc_cn_b : dec_cn_b;
      g.out = enc ? eco[d + 1] : ech[d + 1];
      g.K = 256; g.h = h;
      gemm(g);
    }
  };
  auto bn = [&](int d) {
    int h = (128 >> d) >> 1;
    for (int e = 0; e < 2; ++e) {
      bool enc = (e == 0);
      GemmParams g{};
      setA(g, d, enc);
      g.comp = COMP + d * BLn;
      g.xb = XBUF + xoffE(d + 1);
      g.lab = enc ? enc_lab : dec_lab;
      g.W = wbf + (enc ? 65536 : 114688);
      g.bias = enc ? enc_bn_b : dec_bn_b;
      g.out = enc ? eco[d + 1] : ech[d + 1];
      g.K = 384; g.h = h;
      gemm(g);
    }
  };

  for (int c = 0; c < Nn; ++c) {
    if (c == 0) {
      for (int d = 0; d <= 6; ++d) cn(d);
    } else {
      int z = __builtin_ctz(c);
      int da = 6 - z;
      bn(da);
      for (int d = da + 1; d <= 6; ++d) cn(d);
    }
    int* xdst = (c & 1) ? XCUR : (XBUF + xoffE(7));
    int xstride = (c & 1) ? 128 : 1;
    leaf_kernel<<<Bn, 64, 0, stream>>>(eco[7], ech[7],
        enc_llr_W, enc_llr_b, dec_llr_W, dec_llr_b,
        f, r, PM, PERM, UHAT, PU, PUY, xdst, xstride, c);

    int pendMask = 0, activeMask = 0;
    for (int dd = 1; dd <= 7; ++dd)
      if ((c >> (7 - dd)) & 1) pendMask |= 1 << dd;
    for (int d = 0; d <= 6; ++d)
      if (!((c >> (6 - d)) & 1)) activeMask |= 1 << d;
    perm_apply_kernel<<<128, 256, 0, stream>>>(PERM, UHAT, PU, PUY, XBUF, COMP,
                                               c, pendMask, activeMask);
    if (c & 1) {
      int m = __builtin_ctz(c + 1);
      merge_kernel<<<BLn, 128, 0, stream>>>(XCUR, XBUF, m);
    }
  }
  final_kernel<<<Bn, 128, 0, stream>>>(PM, UHAT, PUY, PU, (float*)d_out);
}
